// Mamba2Forecaster_69432441307447
// MI455X (gfx1250) — matrix operand order repaired and checked by reading, compiled, not run on hardware
//
#include <hip/hip_runtime.h>
#include <hip/hip_bf16.h>

// ---------------- model constants ----------------
#define D_MODEL   768
#define D_INNER   1536
#define D_STATE   128
#define HEADDIM   64
#define NHEADS    24
#define NLAYERS   4
#define DCONV     4
#define CHUNKQ    256
#define CONV_DIM  1792          // D_INNER + 2*D_STATE
#define DIN       3352          // 2*D_INNER + 2*D_STATE + NHEADS
#define INPUT_DIM 32
#define HORIZON   24
#define BATCH     8
#define SEQL      1024
#define NTOK      (BATCH*SEQL)  // 8192
#define NCHUNK    (SEQL/CHUNKQ) // 4
#define NBC       (BATCH*NCHUNK)// 32
#define EPSF      1e-5f

typedef _Float16 f16;
typedef _Float16 v16h __attribute__((ext_vector_type(16)));
typedef _Float16 v8h  __attribute__((ext_vector_type(8)));
typedef float    v8f  __attribute__((ext_vector_type(8)));

// ---------------- WMMA helpers (gfx1250, wave32) ----------------
__device__ __forceinline__ v8f wmma_f16(v16h a, v16h b, v8f c) {
  return __builtin_amdgcn_wmma_f32_16x16x32_f16(
      false, a, false, b, (short)0, c, false, false);
}
__device__ __forceinline__ v8h zero_v8h() { v8h z = {}; return z; }

// A fragment from LDS tile row-major [rows][32] f16.
// ISA: lanes 0-15 row M=lane, K=0..7 & 16..23; lanes 16-31 K=8..15 & 24..31.
__device__ __forceinline__ v16h load_afrag(const f16* Alds, int row0) {
  int l = threadIdx.x & 31;
  const f16* p = Alds + (row0 + (l & 15)) * 32 + ((l >> 4) << 3);
  v16h a;
#pragma unroll
  for (int t = 0; t < 8; ++t) { a[t] = p[t]; a[t + 8] = p[t + 16]; }
  return a;
}
// B fragment from LDS stored transposed [cols][32] f16 (Bt[n][k] = B[k][n]).
// ISA: lanes 0-15 col N=lane, K=0..7 & 16..23; lanes 16-31 K=8..15 & 24..31.
__device__ __forceinline__ v16h load_bfrag(const f16* Bt, int col0) {
  int l = threadIdx.x & 31;
  const f16* p = Bt + (col0 + (l & 15)) * 32 + ((l >> 4) << 3);
  v16h b;
#pragma unroll
  for (int t = 0; t < 8; ++t) { b[t] = p[t]; b[t + 8] = p[t + 16]; }
  return b;
}

__device__ __forceinline__ float siluf(float x) { return x / (1.f + __expf(-x)); }

// ---------------- fill ones ----------------
__global__ void __launch_bounds__(256) k_fill1(float* __restrict__ o, long n) {
  long i = (long)blockIdx.x * 256 + threadIdx.x;
  if (i < n) o[i] = 1.f;
}

// ---------------- transpose-convert: o[N][K] f16 = w[K][N] f32 * s[k] --------
__global__ void __launch_bounds__(256)
k_cvt_t(const float* __restrict__ w, const float* __restrict__ s,
        f16* __restrict__ o, int K, int N) {
  __shared__ f16 tile[32][33];
  int tx = threadIdx.x & 31, ty = threadIdx.x >> 5;   // 32 x 8
  int kbase = blockIdx.y * 32, nbase = blockIdx.x * 32;
#pragma unroll
  for (int i = 0; i < 4; ++i) {
    int k = kbase + ty + i * 8, n = nbase + tx;
    float v = 0.f;
    if (k < K && n < N) v = w[(long)k * N + n] * (s ? s[k] : 1.f);
    tile[ty + i * 8][tx] = (f16)v;
  }
  __syncthreads();
#pragma unroll
  for (int i = 0; i < 4; ++i) {
    int n = nbase + ty + i * 8, k = kbase + tx;
    if (n < N && k < K) o[(long)n * K + k] = tile[tx][ty + i * 8];
  }
}

// ---------------- main WMMA GEMM ----------------
// C[M,N] = (A * rowscale) @ Wt^T (+bias) (+Csrc).  Wt is [N][K] f16.
// Block tile 128x128, BK=32, 8 waves (4 in M x 2 in N), double-buffered LDS.
__global__ void __launch_bounds__(256)
k_gemm(const float* __restrict__ A, const float* __restrict__ rowscale,
       const f16* __restrict__ Wt, float* __restrict__ C,
       const float* __restrict__ Csrc, const float* __restrict__ bias,
       int M, int N, int K, int lda, int ldc) {
  __shared__ f16 Alds[2][128 * 32];
  __shared__ f16 Bt[2][128 * 32];
  const int tid = threadIdx.x, wid = tid >> 5;
  const int wm = wid & 3, wn = wid >> 2;
  const int m0 = blockIdx.y * 128, n0 = blockIdx.x * 128;
  // per-thread staging coords: 16 contiguous elements each for A and Wt
  const int ar = tid >> 1, ac = (tid & 1) * 16;    // A tile 128x32 fp32
  const int bn = tid >> 1, bk = (tid & 1) * 16;    // Wt tile 128x32 f16
  const bool aval = (m0 + ar) < M;
  const bool bval = (n0 + bn) < N;
  const float ascale = aval ? rowscale[m0 + ar] : 0.f;
  const float* Abase = A + (long)(m0 + ar) * lda + ac;
  const f16* Bbase = Wt + (long)(n0 + bn) * K + bk;

  float4 ra[4];
  v8h rb[2];
  v8f acc[8] = {};

  auto loadg = [&](int k0) {
    if (aval) {
      const float4* pa = (const float4*)(Abase + k0);
#pragma unroll
      for (int i = 0; i < 4; ++i) ra[i] = pa[i];
    } else {
#pragma unroll
      for (int i = 0; i < 4; ++i) ra[i] = make_float4(0.f, 0.f, 0.f, 0.f);
    }
    if (bval) {
      const v8h* pb = (const v8h*)(Bbase + k0);
      rb[0] = pb[0]; rb[1] = pb[1];
    } else {
      rb[0] = zero_v8h(); rb[1] = zero_v8h();
    }
  };
  auto stores = [&](int buf) {
    const float* raf = (const float*)ra;
    v8h h0, h1;
#pragma unroll
    for (int i = 0; i < 8; ++i) h0[i] = (f16)(raf[i] * ascale);
#pragma unroll
    for (int i = 0; i < 8; ++i) h1[i] = (f16)(raf[8 + i] * ascale);
    v8h* pa = (v8h*)&Alds[buf][ar * 32 + ac];
    pa[0] = h0; pa[1] = h1;
    v8h* pb = (v8h*)&Bt[buf][bn * 32 + bk];
    pb[0] = rb[0]; pb[1] = rb[1];
  };
  auto compute = [&](int buf) {
    v16h a0 = load_afrag(&Alds[buf][0], wm * 32);
    v16h a1 = load_afrag(&Alds[buf][0], wm * 32 + 16);
#pragma unroll
    for (int nf = 0; nf < 4; ++nf) {
      v16h b = load_bfrag(&Bt[buf][0], wn * 64 + nf * 16);
      acc[nf]     = wmma_f16(a0, b, acc[nf]);
      acc[4 + nf] = wmma_f16(a1, b, acc[4 + nf]);
    }
  };

  loadg(0);
  stores(0);
  __syncthreads();
  int buf = 0;
  for (int k0 = 0; k0 < K; k0 += 32) {
    const bool nxt = (k0 + 32) < K;
    if (nxt) loadg(k0 + 32);
    if (bval && (k0 + 64) < K) __builtin_prefetch(Bbase + k0 + 64, 0, 1);
    compute(buf);
    if (nxt) {
      stores(buf ^ 1);
      __syncthreads();
    }
    buf ^= 1;
  }

  int l = tid & 31, nl = l & 15, mb = (l >> 4) * 8;
#pragma unroll
  for (int af = 0; af < 2; ++af)
#pragma unroll
    for (int nf = 0; nf < 4; ++nf)
#pragma unroll
      for (int e = 0; e < 8; ++e) {
        int gr = m0 + wm * 32 + af * 16 + mb + e;
        int gn = n0 + wn * 64 + nf * 16 + nl;
        if (gr < M && gn < N) {
          float v = acc[af * 4 + nf][e];
          if (bias) v += bias[gn];
          if (Csrc) v += Csrc[(long)gr * ldc + gn];
          C[(long)gr * ldc + gn] = v;
        }
      }
}

// ---------------- per-row rstd ----------------
__global__ void __launch_bounds__(256) k_rstd(const float* __restrict__ x,
                                              float* __restrict__ rstd, int K) {
  long row = blockIdx.x;
  __shared__ float red[256];
  float ss = 0.f;
  for (int c = threadIdx.x; c < K; c += 256) { float v = x[row * K + c]; ss += v * v; }
  red[threadIdx.x] = ss; __syncthreads();
  for (int off = 128; off > 0; off >>= 1) {
    if (threadIdx.x < off) red[threadIdx.x] += red[threadIdx.x + off];
    __syncthreads();
  }
  if (threadIdx.x == 0) rstd[row] = rsqrtf(red[0] / (float)K + EPSF);
}

// ---------------- causal depthwise conv (k=4) + silu ----------------
__global__ void __launch_bounds__(256)
k_conv(const float* __restrict__ zx, const float* __restrict__ cw,
       const float* __restrict__ cb, float* __restrict__ xBC) {
  long row = blockIdx.x;           // b*L + t
  int t = (int)(row & (SEQL - 1));
  for (int c = threadIdx.x; c < CONV_DIM; c += 256) {
    float acc = cb[c];
#pragma unroll
    for (int i = 0; i < DCONV; ++i) {
      int tt = t - (DCONV - 1) + i;
      if (tt >= 0) acc += cw[c * DCONV + i] * zx[(row - (DCONV - 1) + i) * DIN + D_INNER + c];
    }
    xBC[row * CONV_DIM + c] = siluf(acc);
  }
}

// ---------------- dt softplus + per-chunk cumsum of dt*A ----------------
__global__ void __launch_bounds__(256)
k_cumsum(const float* __restrict__ zx, const float* __restrict__ dt_bias,
         const float* __restrict__ A_log, float* __restrict__ dtb,
         float* __restrict__ acum, float* __restrict__ cdecay) {
  int h = blockIdx.x, bc = blockIdx.y;
  int t = threadIdx.x;
  long l = (long)bc * CHUNKQ + t;
  float raw = zx[l * DIN + (D_INNER + CONV_DIM) + h] + dt_bias[h];
  float dtv = (raw > 20.f) ? raw : log1pf(__expf(raw));
  dtb[l * NHEADS + h] = dtv;
  float Aneg = -__expf(A_log[h]);
  __shared__ float s[CHUNKQ];
  s[t] = dtv * Aneg; __syncthreads();
  for (int off = 1; off < CHUNKQ; off <<= 1) {
    float add = (t >= off) ? s[t - off] : 0.f;
    __syncthreads();
    s[t] += add; __syncthreads();
  }
  float cum = s[t];
  acum[((long)bc * NHEADS + h) * CHUNKQ + t] = cum;
  if (t == CHUNKQ - 1) cdecay[(long)bc * NHEADS + h] = __expf(cum);
}

// ---------------- G = Cc @ Bc^T per (b,chunk) ----------------
__global__ void __launch_bounds__(256)
k_G(const float* __restrict__ xBC, float* __restrict__ G) {
  int bc = blockIdx.z;
  int m0 = blockIdx.y * 128, n0 = blockIdx.x * 64;
  long lbase = (long)bc * CHUNKQ;
  __shared__ f16 Alds[128 * 32];
  __shared__ f16 Bt[64 * 32];
  const int tid = threadIdx.x, wid = tid >> 5;
  const int arr = tid >> 1, arc = (tid & 1) * 16;   // A tile 128x32
  const int brn = tid >> 2, brk = (tid & 3) * 8;    // B tile 64x32
  v8f acc[4] = {};
  for (int k0 = 0; k0 < D_STATE; k0 += 32) {
    {  // A = Cc rows (contiguous, vectorized)
      const float4* pa = (const float4*)(xBC + (lbase + m0 + arr) * CONV_DIM +
                                         (D_INNER + D_STATE) + k0 + arc);
      float4 r0 = pa[0], r1 = pa[1], r2 = pa[2], r3 = pa[3];
      const float rf[16] = {r0.x,r0.y,r0.z,r0.w, r1.x,r1.y,r1.z,r1.w,
                            r2.x,r2.y,r2.z,r2.w, r3.x,r3.y,r3.z,r3.w};
      v8h h0, h1;
#pragma unroll
      for (int i = 0; i < 8; ++i) { h0[i] = (f16)rf[i]; h1[i] = (f16)rf[8 + i]; }
      v8h* pl = (v8h*)&Alds[arr * 32 + arc];
      pl[0] = h0; pl[1] = h1;
    }
    {  // Bt[j][k] = Bc[j][k] (already row-contiguous in k)
      const float4* pb = (const float4*)(xBC + (lbase + n0 + brn) * CONV_DIM +
                                         D_INNER + k0 + brk);
      float4 r0 = pb[0], r1 = pb[1];
      const float rf[8] = {r0.x,r0.y,r0.z,r0.w, r1.x,r1.y,r1.z,r1.w};
      v8h h;
#pragma unroll
      for (int i = 0; i < 8; ++i) h[i] = (f16)rf[i];
      *(v8h*)&Bt[brn * 32 + brk] = h;
    }
    __syncthreads();
    v16h a = load_afrag(Alds, wid * 16);
#pragma unroll
    for (int nf = 0; nf < 4; ++nf) {
      v16h b = load_bfrag(Bt, nf * 16);
      acc[nf] = wmma_f16(a, b, acc[nf]);
    }
    __syncthreads();
  }
  int l = tid & 31, nl = l & 15, mb = (l >> 4) * 8;
  float* Gout = G + (long)bc * CHUNKQ * CHUNKQ;
#pragma unroll
  for (int nf = 0; nf < 4; ++nf)
#pragma unroll
    for (int e = 0; e < 8; ++e) {
      int gi = m0 + wid * 16 + mb + e, gj = n0 + nf * 16 + nl;
      Gout[(long)gi * CHUNKQ + gj] = acc[nf][e];
    }
}

// ---------------- Ydiag: Y = (G .* decay-mask) @ xdt per (b,chunk,head) -------
__global__ void __launch_bounds__(256)
k_ydiag(const float* __restrict__ G, const float* __restrict__ acum,
        const float* __restrict__ xBC, const float* __restrict__ dtb,
        float* __restrict__ ybuf) {
  int m0 = blockIdx.x * 128;
  int h = blockIdx.y;
  int bc = blockIdx.z;
  long lbase = (long)bc * CHUNKQ;
  __shared__ f16 Alds[128 * 32];
  __shared__ f16 Bt[64 * 32];
  __shared__ float ac[CHUNKQ];
  const int tid = threadIdx.x, wid = tid >> 5;
  ac[tid] = acum[((long)bc * NHEADS + h) * CHUNKQ + tid];
  __syncthreads();
  v8f acc[4] = {};
  for (int j0 = 0; j0 <= m0 + 127 && j0 < CHUNKQ; j0 += 32) {  // triangular skip
#pragma unroll
    for (int i = 0; i < 16; ++i) {   // masked decayed G tile
      int idx = tid + i * 256; int r = idx >> 5, kk = idx & 31;
      int gi = m0 + r, gj = j0 + kk;
      float v = 0.f;
      if (gi >= gj)
        v = G[((long)bc * CHUNKQ + gi) * CHUNKQ + gj] * __expf(ac[gi] - ac[gj]);
      Alds[r * 32 + kk] = (f16)v;
    }
#pragma unroll
    for (int i = 0; i < 8; ++i) {    // Bt[p][j] = x[j][p]*dt[j]
      int idx = tid + i * 256; int kk = idx >> 6, p = idx & 63;
      long row = lbase + j0 + kk;
      float v = xBC[row * CONV_DIM + h * HEADDIM + p] * dtb[row * NHEADS + h];
      Bt[p * 32 + kk] = (f16)v;
    }
    __syncthreads();
    v16h a = load_afrag(Alds, wid * 16);
#pragma unroll
    for (int nf = 0; nf < 4; ++nf) {
      v16h b = load_bfrag(Bt, nf * 16);
      acc[nf] = wmma_f16(a, b, acc[nf]);
    }
    __syncthreads();
  }
  int l = tid & 31, nl = l & 15, mb = (l >> 4) * 8;
#pragma unroll
  for (int nf = 0; nf < 4; ++nf)
#pragma unroll
    for (int e = 0; e < 8; ++e) {
      int row = m0 + wid * 16 + mb + e, p = nf * 16 + nl;
      ybuf[(lbase + row) * D_INNER + h * HEADDIM + p] = acc[nf][e];
    }
}

// ---------------- chunk states ----------------
__global__ void __launch_bounds__(256)
k_states(const float* __restrict__ xBC, const float* __restrict__ dtb,
         const float* __restrict__ acum, float* __restrict__ states) {
  int h = blockIdx.x, bc = blockIdx.y;
  long lbase = (long)bc * CHUNKQ;
  const float* ac = acum + ((long)bc * NHEADS + h) * CHUNKQ;
  __shared__ f16 Alds[64 * 32];
  __shared__ f16 Bt[128 * 32];
  const int tid = threadIdx.x, wid = tid >> 5;
  const int wm = wid & 3, wn = wid >> 2;
  float aclast = ac[CHUNKQ - 1];
  v8f acc[4] = {};
  for (int j0 = 0; j0 < CHUNKQ; j0 += 32) {
#pragma unroll
    for (int i = 0; i < 8; ++i) {    // A[p][j] = x[j][p]*dt[j]*decay[j]
      int idx = tid + i * 256; int kk = idx >> 6, p = idx & 63;
      long row = lbase + j0 + kk;
      float dec = __expf(aclast - ac[j0 + kk]);
      float v = xBC[row * CONV_DIM + h * HEADDIM + p] * dtb[row * NHEADS + h] * dec;
      Alds[p * 32 + kk] = (f16)v;
    }
#pragma unroll
    for (int i = 0; i < 16; ++i) {   // Bt[n][j] = Bc[j][n]
      int idx = tid + i * 256; int kk = idx >> 7, n = idx & 127;
      long row = lbase + j0 + kk;
      Bt[n * 32 + kk] = (f16)xBC[row * CONV_DIM + D_INNER + n];
    }
    __syncthreads();
    v16h a = load_afrag(Alds, wm * 16);
#pragma unroll
    for (int nf = 0; nf < 4; ++nf) {
      v16h b = load_bfrag(Bt, wn * 64 + nf * 16);
      acc[nf] = wmma_f16(a, b, acc[nf]);
    }
    __syncthreads();
  }
  int l = tid & 31, nl = l & 15, mb = (l >> 4) * 8;
  float* S = states + ((long)bc * NHEADS + h) * (HEADDIM * D_STATE);
#pragma unroll
  for (int nf = 0; nf < 4; ++nf)
#pragma unroll
    for (int e = 0; e < 8; ++e) {
      int p = wm * 16 + mb + e, n = wn * 64 + nf * 16 + nl;
      S[p * D_STATE + n] = acc[nf][e];
    }
}

// ---------------- sequential inter-chunk scan ----------------
__global__ void __launch_bounds__(256)
k_scan(const float* __restrict__ states, const float* __restrict__ cdecay,
       float* __restrict__ prev) {
  long idx = (long)blockIdx.x * 256 + threadIdx.x;
  const long total = (long)BATCH * NHEADS * HEADDIM * D_STATE;
  if (idx >= total) return;
  int pn = (int)(idx & (HEADDIM * D_STATE - 1));
  int h = (int)((idx >> 13) % NHEADS);
  int b = (int)(idx / ((long)NHEADS * HEADDIM * D_STATE));
  float s = 0.f;
  for (int c = 0; c < NCHUNK; ++c) {
    long bc = (long)b * NCHUNK + c;
    long o = (bc * NHEADS + h) * (HEADDIM * D_STATE) + pn;
    prev[o] = s;
    s = states[o] + cdecay[bc * NHEADS + h] * s;
  }
}

// ---------------- Yoff: y += exp(Acum)*(Cc @ prev^T) + x*D ----------------
__global__ void __launch_bounds__(256)
k_yoff(const float* __restrict__ xBC, const float* __restrict__ acum,
       const float* __restrict__ prev, const float* __restrict__ Dw,
       float* __restrict__ ybuf) {
  int m0 = blockIdx.x * 128;
  int h = blockIdx.y;
  int bc = blockIdx.z;
  long lbase = (long)bc * CHUNKQ;
  const float* ac = acum + ((long)bc * NHEADS + h) * CHUNKQ;
  const float* P = prev + ((long)bc * NHEADS + h) * (HEADDIM * D_STATE);
  __shared__ f16 Alds[128 * 32];
  __shared__ f16 Bt[64 * 32];
  const int tid = threadIdx.x, wid = tid >> 5;
  const int arr = tid >> 1, arc = (tid & 1) * 16;   // A 128x32
  const int brp = tid >> 2, brk = (tid & 3) * 8;    // B 64x32
  v8f acc[4] = {};
  for (int k0 = 0; k0 < D_STATE; k0 += 32) {
    {  // A = Cc (vectorized)
      const float4* pa = (const float4*)(xBC + (lbase + m0 + arr) * CONV_DIM +
                                         (D_INNER + D_STATE) + k0 + arc);
      float4 r0 = pa[0], r1 = pa[1], r2 = pa[2], r3 = pa[3];
      const float rf[16] = {r0.x,r0.y,r0.z,r0.w, r1.x,r1.y,r1.z,r1.w,
                            r2.x,r2.y,r2.z,r2.w, r3.x,r3.y,r3.z,r3.w};
      v8h h0, h1;
#pragma unroll
      for (int i = 0; i < 8; ++i) { h0[i] = (f16)rf[i]; h1[i] = (f16)rf[8 + i]; }
      v8h* pl = (v8h*)&Alds[arr * 32 + arc];
      pl[0] = h0; pl[1] = h1;
    }
    {  // Bt[p][n] = prev[p][n] (row-contiguous)
      const float4* pb = (const float4*)(P + brp * D_STATE + k0 + brk);
      float4 r0 = pb[0], r1 = pb[1];
      const float rf[8] = {r0.x,r0.y,r0.z,r0.w, r1.x,r1.y,r1.z,r1.w};
      v8h hh;
#pragma unroll
      for (int i = 0; i < 8; ++i) hh[i] = (f16)rf[i];
      *(v8h*)&Bt[brp * 32 + brk] = hh;
    }
    __syncthreads();
    v16h a = load_afrag(Alds, wid * 16);
#pragma unroll
    for (int nf = 0; nf < 4; ++nf) {
      v16h b = load_bfrag(Bt, nf * 16);
      acc[nf] = wmma_f16(a, b, acc[nf]);
    }
    __syncthreads();
  }
  int l = tid & 31, nl = l & 15, mb = (l >> 4) * 8;
#pragma unroll
  for (int nf = 0; nf < 4; ++nf)
#pragma unroll
    for (int e = 0; e < 8; ++e) {
      int row = m0 + wid * 16 + mb + e, p = nf * 16 + nl;
      long lrow = lbase + row;
      float xs = xBC[lrow * CONV_DIM + h * HEADDIM + p];
      ybuf[lrow * D_INNER + h * HEADDIM + p] +=
          acc[nf][e] * __expf(ac[row]) + xs * Dw[h];
    }
}

// ---------------- gate: g = y*silu(z) + row rstd ----------------
__global__ void __launch_bounds__(256)
k_gate_rstd(const float* __restrict__ zx, const float* __restrict__ ybuf,
            float* __restrict__ g, float* __restrict__ rstd) {
  long row = blockIdx.x;
  __shared__ float red[256];
  float ss = 0.f;
  for (int c = threadIdx.x; c < D_INNER; c += 256) {
    float z = zx[row * DIN + c];
    float gg = ybuf[row * D_INNER + c] * siluf(z);
    g[row * D_INNER + c] = gg;
    ss += gg * gg;
  }
  red[threadIdx.x] = ss; __syncthreads();
  for (int off = 128; off > 0; off >>= 1) {
    if (threadIdx.x < off) red[threadIdx.x] += red[threadIdx.x + off];
    __syncthreads();
  }
  if (threadIdx.x == 0) rstd[row] = rsqrtf(red[0] / (float)D_INNER + EPSF);
}

// ---------------- final head ----------------
__global__ void __launch_bounds__(256)
k_head(const float* __restrict__ hbuf, const float* __restrict__ nw,
       const float* __restrict__ ow, const float* __restrict__ ob,
       float* __restrict__ out) {
  int b = blockIdx.x;
  const float* hr = hbuf + ((long)b * SEQL + (SEQL - 1)) * D_MODEL;
  __shared__ float red[256];
  __shared__ float hn[D_MODEL];
  __shared__ float rs;
  int tid = threadIdx.x;
  float ss = 0.f;
  for (int c = tid; c < D_MODEL; c += 256) { float v = hr[c]; ss += v * v; }
  red[tid] = ss; __syncthreads();
  for (int off = 128; off > 0; off >>= 1) {
    if (tid < off) red[tid] += red[tid + off];
    __syncthreads();
  }
  if (tid == 0) rs = rsqrtf(red[0] / (float)D_MODEL + EPSF);
  __syncthreads();
  for (int c = tid; c < D_MODEL; c += 256) hn[c] = hr[c] * rs * nw[c];
  __syncthreads();
  int wid = tid >> 5, l = tid & 31;
#pragma unroll
  for (int rep = 0; rep < 3; ++rep) {
    int n = wid + rep * 8;
    float acc = 0.f;
    for (int k = l; k < D_MODEL; k += 32) acc += hn[k] * ow[k * HORIZON + n];
#pragma unroll
    for (int off = 16; off > 0; off >>= 1) acc += __shfl_down(acc, off, 32);
    if (l == 0) out[b * HORIZON + n] = acc + ob[n];
  }
}

// =====================================================================
extern "C" void kernel_launch(void* const* d_in, const int* in_sizes, int n_in,
                              void* d_out, int out_size, void* d_ws, size_t ws_size,
                              hipStream_t stream) {
  (void)in_sizes; (void)n_in; (void)out_size; (void)ws_size;
  const float* x         = (const float*)d_in[0];
  const float* inp_w     = (const float*)d_in[1];
  const float* inp_b     = (const float*)d_in[2];
  const float* in_proj_w = (const float*)d_in[3];
  const float* conv_w    = (const float*)d_in[4];
  const float* conv_b    = (const float*)d_in[5];
  const float* dt_bias   = (const float*)d_in[6];
  const float* A_log     = (const float*)d_in[7];
  const float* Dw        = (const float*)d_in[8];
  const float* gnorm_w   = (const float*)d_in[9];
  const float* out_proj_w= (const float*)d_in[10];
  const float* ln_w      = (const float*)d_in[11];
  const float* norm_f_w  = (const float*)d_in[12];
  const float* outp_w    = (const float*)d_in[13];
  const float* outp_b    = (const float*)d_in[14];
  float* out = (float*)d_out;

  char* wsp = (char*)d_ws;
  auto carve = [&](size_t bytes) -> void* {
    void* p = (void*)wsp;
    wsp += (bytes + 255) & ~(size_t)255;
    return p;
  };
  float* hb     = (float*)carve((size_t)NTOK * D_MODEL * 4);
  float* zx     = (float*)carve((size_t)NTOK * DIN * 4);
  float* xBC    = (float*)carve((size_t)NTOK * CONV_DIM * 4);
  float* dtb    = (float*)carve((size_t)NTOK * NHEADS * 4);
  float* acum   = (float*)carve((size_t)NBC * NHEADS * CHUNKQ * 4);
  float* cdecay = (float*)carve((size_t)NBC * NHEADS * 4);
  float* Gbuf   = (float*)carve((size_t)NBC * CHUNKQ * CHUNKQ * 4);
  float* ybuf   = (float*)carve((size_t)NTOK * D_INNER * 4);
  float* states = (float*)carve((size_t)NBC * NHEADS * HEADDIM * D_STATE * 4);
  float* prevb  = (float*)carve((size_t)NBC * NHEADS * HEADDIM * D_STATE * 4);
  float* gbuf   = (float*)carve((size_t)NTOK * D_INNER * 4);
  float* rstd   = (float*)carve((size_t)NTOK * 4);
  float* ones   = (float*)carve((size_t)NTOK * 4);
  f16* wt_inp     = (f16*)carve((size_t)D_MODEL * INPUT_DIM * 2);           // [768][32]
  f16* wt_inproj  = (f16*)carve((size_t)NLAYERS * DIN * D_MODEL * 2);       // [3352][768]
  f16* wt_outproj = (f16*)carve((size_t)NLAYERS * D_MODEL * D_INNER * 2);   // [768][1536]

  k_fill1<<<(NTOK + 255) / 256, 256, 0, stream>>>(ones, NTOK);

  // ---- weight conversion to transposed f16 [N][K] (rms weights folded) ----
  {
    dim3 grid((D_MODEL + 31) / 32, (INPUT_DIM + 31) / 32);
    k_cvt_t<<<grid, 256, 0, stream>>>(inp_w, nullptr, wt_inp, INPUT_DIM, D_MODEL);
  }
  for (int l = 0; l < NLAYERS; ++l) {
    {
      dim3 grid((DIN + 31) / 32, (D_MODEL + 31) / 32);
      k_cvt_t<<<grid, 256, 0, stream>>>(in_proj_w + (long)l * D_MODEL * DIN,
                                        ln_w + (long)l * D_MODEL,
                                        wt_inproj + (long)l * DIN * D_MODEL,
                                        D_MODEL, DIN);
    }
    {
      dim3 grid((D_MODEL + 31) / 32, (D_INNER + 31) / 32);
      k_cvt_t<<<grid, 256, 0, stream>>>(out_proj_w + (long)l * D_INNER * D_MODEL,
                                        gnorm_w + (long)l * D_INNER,
                                        wt_outproj + (long)l * D_MODEL * D_INNER,
                                        D_INNER, D_MODEL);
    }
  }

  // ---- input projection: h = x @ inp_w + inp_b ----
  {
    dim3 grid((D_MODEL + 127) / 128, (NTOK + 127) / 128);
    k_gemm<<<grid, 256, 0, stream>>>(x, ones, wt_inp, hb, nullptr, inp_b,
                                     NTOK, D_MODEL, INPUT_DIM, INPUT_DIM, D_MODEL);
  }

  for (int l = 0; l < NLAYERS; ++l) {
    k_rstd<<<NTOK, 256, 0, stream>>>(hb, rstd, D_MODEL);
    {
      dim3 grid((DIN + 127) / 128, (NTOK + 127) / 128);
      k_gemm<<<grid, 256, 0, stream>>>(hb, rstd, wt_inproj + (long)l * DIN * D_MODEL,
                                       zx, nullptr, nullptr,
                                       NTOK, DIN, D_MODEL, D_MODEL, DIN);
    }
    k_conv<<<NTOK, 256, 0, stream>>>(zx, conv_w + (long)l * CONV_DIM * DCONV,
                                     conv_b + (long)l * CONV_DIM, xBC);
    {
      dim3 grid(NHEADS, NBC);
      k_cumsum<<<grid, 256, 0, stream>>>(zx, dt_bias + (long)l * NHEADS,
                                         A_log + (long)l * NHEADS, dtb, acum, cdecay);
    }
    {
      dim3 grid(CHUNKQ / 64, CHUNKQ / 128, NBC);
      k_G<<<grid, 256, 0, stream>>>(xBC, Gbuf);
    }
    {
      dim3 grid(CHUNKQ / 128, NHEADS, NBC);
      k_ydiag<<<grid, 256, 0, stream>>>(Gbuf, acum, xBC, dtb, ybuf);
    }
    {
      dim3 grid(NHEADS, NBC);
      k_states<<<grid, 256, 0, stream>>>(xBC, dtb, acum, states);
    }
    {
      long total = (long)BATCH * NHEADS * HEADDIM * D_STATE;
      k_scan<<<(unsigned)((total + 255) / 256), 256, 0, stream>>>(states, cdecay, prevb);
    }
    {
      dim3 grid(CHUNKQ / 128, NHEADS, NBC);
      k_yoff<<<grid, 256, 0, stream>>>(xBC, acum, prevb, Dw + (long)l * NHEADS, ybuf);
    }
    k_gate_rstd<<<NTOK, 256, 0, stream>>>(zx, ybuf, gbuf, rstd);
    {
      dim3 grid((D_MODEL + 127) / 128, (NTOK + 127) / 128);
      k_gemm<<<grid, 256, 0, stream>>>(gbuf, rstd,
                                       wt_outproj + (long)l * D_MODEL * D_INNER,
                                       hb, hb, nullptr,
                                       NTOK, D_MODEL, D_INNER, D_INNER, D_MODEL);
    }
  }

  k_head<<<BATCH, 256, 0, stream>>>(hb, norm_f_w, outp_w, outp_b, out);
}
